// MSMamba_2104533975582
// MI455X (gfx1250) — compile-verified, matrix operand order checked by reading, not run on hardware
//
#include <hip/hip_runtime.h>
#include <hip/hip_bf16.h>
#include <math.h>

// ---------------------------------------------------------------------------
// Mamba forward for MI455X (gfx1250, wave32, WMMA + TDM).
//  - GEMMs: v_wmma_f32_16x16x32_bf16, split-bf16 (hi+lo) for ~fp32 accuracy,
//    32x32 macro-tile per wave (4 accumulators, 12 WMMA per k-chunk).
//  - Scan: per-(b,d) channel in registers; shared B/C stream staged into LDS
//    by the Tensor Data Mover (tensor_load_to_lds), double-buffered,
//    synchronized with s_wait_tensorcnt.
// ---------------------------------------------------------------------------

#define B_SZ    4
#define L_SZ    2048
#define D_MODEL 1024
#define D_INNER 2048
#define N_STATE 16
#define R_RANK  64
#define NBC     96            // dt_rank + 2*d_state
#define CH      64            // scan l-chunk staged per TDM transfer
#define NCHUNK  (L_SZ / CH)   // 32

typedef __attribute__((ext_vector_type(16))) __bf16 v16bf;
typedef __attribute__((ext_vector_type(8)))  float  v8f;
typedef __attribute__((ext_vector_type(4)))  unsigned int u32x4;
typedef __attribute__((ext_vector_type(8)))  int i32x8;
typedef __attribute__((ext_vector_type(4)))  int i32x4;

// ---------------------------------------------------------------------------
// TDM: DMA a 2D tile (rows x width fp32, row stride in elements) from global
// memory into LDS. D# bitfields per CDNA5 ISA ch.8 (group0/group1; groups
// 2/3 zero for a 2D tensor). Issue from one wave only; track with TENSORcnt.
// ---------------------------------------------------------------------------
__device__ __forceinline__ void tdm_load_2d_to_lds(
    const void* gptr, unsigned lds_off,
    unsigned width_elts, unsigned rows, unsigned row_stride_elts) {
  unsigned long long ga = (unsigned long long)(uintptr_t)gptr;
  u32x4 g0;
  g0[0] = 1u;                                    // [1:0] count=1 (user mode)
  g0[1] = lds_off;                               // [63:32] lds_addr (bytes)
  g0[2] = (unsigned)ga;                          // [95:64] global_addr lo
  g0[3] = ((unsigned)(ga >> 32) & 0x01FFFFFFu)   // [120:96] global_addr hi
        | (2u << 30);                            // [127:126] type=2 ("image")
  i32x8 g1;
  g1[0] = (int)(2u << 16);                       // data_size=2 -> 4-byte elts
  g1[1] = (int)((width_elts & 0xFFFFu) << 16);   // tensor_dim0[15:0] @63:48
  g1[2] = (int)((width_elts >> 16) | ((rows & 0xFFFFu) << 16)); // dim0 hi|dim1 lo
  g1[3] = (int)((rows >> 16) | ((width_elts & 0xFFFFu) << 16)); // dim1 hi|tile_dim0
  g1[4] = (int)(rows & 0xFFFFu);                 // tile_dim1 (tile_dim2=0)
  g1[5] = (int)row_stride_elts;                  // tensor_dim0_stride[31:0]
  g1[6] = 0;                                     // stride0 hi | stride1 lo
  g1[7] = 0;                                     // stride1 hi (unused, 2D)
  i32x4 gz = {0, 0, 0, 0};
#if defined(__clang_major__) && __clang_major__ >= 23
  i32x8 gz8 = {0, 0, 0, 0, 0, 0, 0, 0};
  __builtin_amdgcn_tensor_load_to_lds(g0, g1, gz, gz, gz8, 0);
#else
  __builtin_amdgcn_tensor_load_to_lds(g0, g1, gz, gz, 0);
#endif
}

// ---------------------------------------------------------------------------
// Load a 16x32 fp32 tile fragment (rows row0..row0+15, cols k0..k0+31, row
// stride ld) into the ISA 16-bit A/B fragment layout (7.12.2), split into
// bf16 hi + bf16 residual lo. Per lane: row = lane&15, half = lane>>4;
// elements 0..7 hold K = half*8 + e, elements 8..15 hold K = 16 + half*8 + e.
// Each lane's data is two 8-float contiguous runs -> 4x b128 loads.
// ---------------------------------------------------------------------------
__device__ __forceinline__ void load_frag_split(
    const float* __restrict__ base, size_t ld, int row0, int k0, int lane,
    v16bf& hi, v16bf& lo) {
  const int r    = row0 + (lane & 15);
  const int half = (lane >> 4) & 1;
  const float4* p =
      (const float4*)(base + (size_t)r * ld + (size_t)k0 + (size_t)half * 8);
  float4 q0 = p[0], q1 = p[1], q2 = p[4], q3 = p[5];
  float v[16] = {q0.x, q0.y, q0.z, q0.w, q1.x, q1.y, q1.z, q1.w,
                 q2.x, q2.y, q2.z, q2.w, q3.x, q3.y, q3.z, q3.w};
#pragma unroll
  for (int e = 0; e < 16; ++e) {
    __bf16 h = (__bf16)v[e];
    hi[e] = h;
    lo[e] = (__bf16)(v[e] - (float)h);
  }
}

// split-bf16 fp32 emulation: a*b ~= ah*bh + ah*bl + al*bh
__device__ __forceinline__ v8f wmma3(v16bf ah, v16bf al, v16bf bh, v16bf bl,
                                     v8f acc) {
  acc = __builtin_amdgcn_wmma_f32_16x16x32_bf16(false, ah, false, bh,
                                                (short)0, acc, false, false);
  acc = __builtin_amdgcn_wmma_f32_16x16x32_bf16(false, ah, false, bl,
                                                (short)0, acc, false, false);
  acc = __builtin_amdgcn_wmma_f32_16x16x32_bf16(false, al, false, bh,
                                                (short)0, acc, false, false);
  return acc;
}

template<bool SOFTPLUS>
__device__ __forceinline__ void store_tile(
    float* __restrict__ C, size_t ldc, int tm, int tn, int lane, v8f acc,
    const float* __restrict__ bias) {
  // C/D layout: n = lane&15; VGPR r -> m = r + 8*(lane>>4).
  const int n    = tn + (lane & 15);
  const int half = (lane >> 4) & 1;
  float bv = SOFTPLUS ? bias[n] : 0.f;
#pragma unroll
  for (int r = 0; r < 8; ++r) {
    int m = tm + r + half * 8;
    float v = acc[r];
    if (SOFTPLUS) {
      v += bv;
      v = (v > 20.f) ? v : log1pf(__expf(v));    // softplus
    }
    C[(size_t)m * ldc + n] = v;
  }
}

// ---------------------------------------------------------------------------
// C[M,N] = A[M,K] * Wt[N,K]^T   (Wt row-major: Wt[n*ldb + k])
// One 32x32 macro-tile per wave (2x2 WMMA tiles); 8 waves per block.
// SOFTPLUS epilogue: C = softplus(C + bias[n]).
// ---------------------------------------------------------------------------
template<bool SOFTPLUS>
__global__ void __launch_bounds__(256)
mamba_wmma_gemm(const float* __restrict__ A,  int lda,
                const float* __restrict__ Wt, int ldb,
                float* __restrict__ C,        int ldc,
                int M, int N, int K,
                const float* __restrict__ bias) {
  const int lane = threadIdx.x & 31;
  const int wave = threadIdx.x >> 5;
  const int tilesN = N >> 5;                       // 32-wide macro tiles
  const int tile = blockIdx.x * 8 + wave;
  if (tile >= (M >> 5) * tilesN) return;           // wave-uniform branch
  const int tm = (tile / tilesN) << 5;
  const int tn = (tile % tilesN) << 5;

  v8f a00 = {}, a01 = {}, a10 = {}, a11 = {};
  for (int k0 = 0; k0 < K; k0 += 32) {
    v16bf ah0, al0, ah1, al1, bh0, bl0, bh1, bl1;
    load_frag_split(A,  (size_t)lda, tm,      k0, lane, ah0, al0);
    load_frag_split(A,  (size_t)lda, tm + 16, k0, lane, ah1, al1);
    load_frag_split(Wt, (size_t)ldb, tn,      k0, lane, bh0, bl0);
    load_frag_split(Wt, (size_t)ldb, tn + 16, k0, lane, bh1, bl1);
    a00 = wmma3(ah0, al0, bh0, bl0, a00);
    a01 = wmma3(ah0, al0, bh1, bl1, a01);
    a10 = wmma3(ah1, al1, bh0, bl0, a10);
    a11 = wmma3(ah1, al1, bh1, bl1, a11);
  }
  store_tile<SOFTPLUS>(C, (size_t)ldc, tm,      tn,      lane, a00, bias);
  store_tile<SOFTPLUS>(C, (size_t)ldc, tm,      tn + 16, lane, a01, bias);
  store_tile<SOFTPLUS>(C, (size_t)ldc, tm + 16, tn,      lane, a10, bias);
  store_tile<SOFTPLUS>(C, (size_t)ldc, tm + 16, tn + 16, lane, a11, bias);
}

// ---------------------------------------------------------------------------
// Depthwise causal conv1d (k=4, left pad 3) + bias + SiLU.
// x_pre, xconv layout (B, L, Di) with Di contiguous -> coalesced in d.
// ---------------------------------------------------------------------------
__global__ void __launch_bounds__(256)
mamba_conv_silu(const float* __restrict__ x_pre,
                const float* __restrict__ cw,   // (Di, 1, 4)
                const float* __restrict__ cb,   // (Di,)
                float* __restrict__ xconv) {
  const size_t idx = (size_t)blockIdx.x * 256 + threadIdx.x;
  const int d = (int)(idx % D_INNER);
  const int l = (int)((idx / D_INNER) % L_SZ);
  const int b = (int)(idx / ((size_t)D_INNER * L_SZ));
  const float* xb = x_pre + (size_t)b * L_SZ * D_INNER;

  float acc = cb[d];
#pragma unroll
  for (int j = 0; j < 4; ++j) {
    int ls = l - 3 + j;
    if (ls >= 0)
      acc += cw[d * 4 + j] * xb[(size_t)ls * D_INNER + d];
  }
  float sig = 1.f / (1.f + __expf(-acc));
  xconv[idx] = acc * sig;
}

// ---------------------------------------------------------------------------
// Selective scan: one thread per (b, d) channel; 16 states in registers.
// The shared B/C operand stream (dtBC cols 64..95) is DMA'd to LDS by the
// TDM in CH-row chunks, double buffered. Fused skip + SiLU(z) gate.
// ---------------------------------------------------------------------------
__global__ void __launch_bounds__(256)
mamba_scan(const float* __restrict__ xconv,    // (B, L, Di)
           const float* __restrict__ dt_full,  // (B, L, Di)  softplus'd
           const float* __restrict__ dtBC,     // (B, L, 96)
           const float* __restrict__ A_log,    // (Di, 16)
           const float* __restrict__ Dp,       // (Di,)
           const float* __restrict__ z,        // (B, Di, L)
           float* __restrict__ out) {          // (B, L, Di)
  __shared__ float bcs[2][CH][32];             // [buf][row][B0..15 C0..15]
  const int d = blockIdx.x * 256 + threadIdx.x;   // 0..Di-1
  const int b = blockIdx.y;

  float An[N_STATE];
#pragma unroll
  for (int n = 0; n < N_STATE; ++n)
    An[n] = -__expf(A_log[(size_t)d * N_STATE + n]);
  float s[N_STATE];
#pragma unroll
  for (int n = 0; n < N_STATE; ++n) s[n] = 0.f;

  const float Dd = Dp[d];
  const float* xb  = xconv   + (size_t)b * L_SZ * D_INNER + d;
  const float* dtb = dt_full + (size_t)b * L_SZ * D_INNER + d;
  const float* bcb = dtBC    + (size_t)b * L_SZ * NBC + 64;  // B/C columns
  const float* zb  = z + ((size_t)b * D_INNER + d) * L_SZ;
  float*       ob  = out + (size_t)b * L_SZ * D_INNER + d;

  const unsigned lds_base = (unsigned)(uintptr_t)&bcs[0][0][0];
  const unsigned buf_bytes = CH * 32 * sizeof(float);

  if (threadIdx.x < 32)                        // wave 0 drives the TDM
    tdm_load_2d_to_lds(bcb, lds_base, 32, CH, NBC);

  for (int c = 0; c < NCHUNK; ++c) {
    if (threadIdx.x < 32) {
      if (c + 1 < NCHUNK) {
        tdm_load_2d_to_lds(bcb + (size_t)(c + 1) * CH * NBC,
                           lds_base + ((unsigned)(c + 1) & 1u) * buf_bytes,
                           32, CH, NBC);
        __builtin_amdgcn_s_wait_tensorcnt(1);  // chunk c landed in LDS
      } else {
        __builtin_amdgcn_s_wait_tensorcnt(0);
      }
    }
    __syncthreads();                           // publish LDS to all waves
    const float (*bc)[32] = bcs[c & 1];
    const int lbase = c * CH;
    for (int j = 0; j < CH; ++j) {
      const int l = lbase + j;
      const float dt = dtb[(size_t)l * D_INNER];
      const float u  = xb[(size_t)l * D_INNER];
      const float du = dt * u;
      float y = 0.f;
#pragma unroll
      for (int n = 0; n < N_STATE; ++n) {
        float dA = __expf(dt * An[n]);         // v_exp_f32 (TRANS)
        s[n] = s[n] * dA + du * bc[j][n];      // ds_load broadcast
        y += s[n] * bc[j][16 + n];
      }
      float zv = zb[l];
      float g = zv / (1.f + __expf(-zv));      // silu(z)
      ob[(size_t)l * D_INNER] = (y + Dd * u) * g;
    }
    __syncthreads();                           // before buffer reuse
  }
}

extern "C" void kernel_launch(void* const* d_in, const int* in_sizes, int n_in,
                              void* d_out, int out_size, void* d_ws, size_t ws_size,
                              hipStream_t stream) {
  const float* hidden    = (const float*)d_in[0];  // (B, L, Dm)
  const float* z         = (const float*)d_in[1];  // (B, Di, L)
  const float* in_proj_w = (const float*)d_in[2];  // (Di, Dm)
  const float* conv_w    = (const float*)d_in[3];  // (Di, 1, 4)
  const float* conv_b    = (const float*)d_in[4];  // (Di,)
  const float* x_proj_w  = (const float*)d_in[5];  // (96, Di)
  const float* dt_proj_w = (const float*)d_in[6];  // (Di, 64)
  const float* dt_proj_b = (const float*)d_in[7];  // (Di,)
  const float* A_log     = (const float*)d_in[8];  // (Di, 16)
  const float* Dp        = (const float*)d_in[9];  // (Di,)
  float* out = (float*)d_out;

  const size_t XBYTES = (size_t)B_SZ * L_SZ * D_INNER * sizeof(float); // 64 MiB
  char* ws = (char*)d_ws;
  float* x_pre   = (float*)ws;                   // (B, L, Di)
  float* xconv   = (float*)(ws + XBYTES);        // (B, L, Di)
  float* dtBC    = (float*)(ws + 2 * XBYTES);    // (B*L, 96)   ~3 MiB
  float* dt_full = x_pre;                        // reuse: x_pre dead after conv

  const int M = B_SZ * L_SZ;                     // 8192 rows (b,l)

  // 1) in_proj: x_pre[(b,l), e] = sum_d hidden[(b,l), d] * in_proj_w[e, d]
  {
    int tiles = (M / 32) * (D_INNER / 32);       // 16384 macro-tiles
    mamba_wmma_gemm<false><<<dim3((tiles + 7) / 8), dim3(256), 0, stream>>>(
        hidden, D_MODEL, in_proj_w, D_MODEL, x_pre, D_INNER,
        M, D_INNER, D_MODEL, nullptr);
  }

  // 2) depthwise conv + bias + SiLU
  {
    size_t elems = (size_t)B_SZ * L_SZ * D_INNER;
    mamba_conv_silu<<<dim3((unsigned)(elems / 256)), dim3(256), 0, stream>>>(
        x_pre, conv_w, conv_b, xconv);
  }

  // 3) x_proj: dtBC[(b,l), e] = sum_d xconv[(b,l), d] * x_proj_w[e, d]
  {
    int tiles = (M / 32) * (NBC / 32);           // 768 macro-tiles
    mamba_wmma_gemm<false><<<dim3((tiles + 7) / 8), dim3(256), 0, stream>>>(
        xconv, D_INNER, x_proj_w, D_INNER, dtBC, NBC,
        M, NBC, D_INNER, nullptr);
  }

  // 4) dt_proj + softplus: dt_full[(b,l), d] =
  //       softplus( sum_r dtBC[(b,l), r] * dt_proj_w[d, r] + dt_proj_b[d] )
  {
    int tiles = (M / 32) * (D_INNER / 32);       // 16384 macro-tiles
    mamba_wmma_gemm<true><<<dim3((tiles + 7) / 8), dim3(256), 0, stream>>>(
        dtBC, NBC, dt_proj_w, R_RANK, dt_full, D_INNER,
        M, D_INNER, R_RANK, dt_proj_b);
  }

  // 5) selective scan + skip + gate, writes final output (B, L, Di)
  {
    mamba_scan<<<dim3(D_INNER / 256, B_SZ), dim3(256), 0, stream>>>(
        xconv, dt_full, dtBC, A_log, Dp, z, out);
  }
}
